// ToyOrthoRotCell_50216757625130
// MI455X (gfx1250) — matrix operand order repaired and checked by reading, not compiled, not run on hardware
//
#include <hip/hip_runtime.h>
#include <hip/hip_bf16.h>

// ---- problem dims ----
constexpr int MDIM = 16384;  // B (batch rows)
constexpr int NDIM = 2048;   // D (output cols)
constexpr int KDIM = 2048;   // I == D (reduction length)

// ---- tiling ----
constexpr int BM = 128;      // workgroup M tile
constexpr int BN = 64;       // workgroup N tile
constexpr int BK = 32;       // K block (one bf16 WMMA K-step)
constexpr int SAB = 40;      // LDS row stride (bf16 elems) for A tiles (80B rows)
constexpr int SBB = 40;      // LDS row stride (bf16 elems) for B tiles

// LDS plane offsets (ushort elems) within one buffer
constexpr int TA = BM * SAB;                 // 5120
constexpr int TB = BN * SBB;                 // 2560
constexpr int OXH = 0, OXL = TA, OHH = 2 * TA, OHL = 3 * TA;
constexpr int OB1H = 4 * TA, OB1L = OB1H + TB;
constexpr int OB2H = OB1H + 2 * TB, OB2L = OB1H + 3 * TB;
constexpr int OB3H = OB1H + 4 * TB, OB3L = OB1H + 5 * TB;
constexpr int BUFE = 4 * TA + 6 * TB;        // 35840 elems = 71680 B / buffer

typedef __attribute__((ext_vector_type(16))) __bf16 v16bf;
typedef __attribute__((ext_vector_type(8)))  __bf16 v8bf;
typedef __attribute__((ext_vector_type(8)))  float  v8f;

__device__ __forceinline__ v8f wmma_bf16(v16bf a, v16bf b, v8f c) {
    return __builtin_amdgcn_wmma_f32_16x16x32_bf16(
        false, a, false, b, (short)0, c, false, false);
}

// ---- bf16 split helpers (RNE) ----
__device__ __forceinline__ unsigned short f2bf(float f) {
    unsigned u = __float_as_uint(f);
    u += 0x7FFFu + ((u >> 16) & 1u);
    return (unsigned short)(u >> 16);
}
__device__ __forceinline__ float bf2f(unsigned short s) {
    return __uint_as_float((unsigned)s << 16);
}
__device__ __forceinline__ void split_bf16(float f, unsigned short& hi, unsigned short& lo) {
    hi = f2bf(f);
    lo = f2bf(f - bf2f(hi));
}
__device__ __forceinline__ v16bf combine16(v8bf a, v8bf b) {
    return __builtin_shufflevector(a, b, 0, 1, 2, 3, 4, 5, 6, 7,
                                   8, 9, 10, 11, 12, 13, 14, 15);
}

// ---- CDNA5 async global->LDS copy (16B per lane), tracked by ASYNCcnt ----
__device__ __forceinline__ void cp16(unsigned short* dst_lds, const unsigned short* src) {
    // generic pointer to an LDS object: low 32 bits are the LDS byte offset
    const unsigned l = (unsigned)(uintptr_t)dst_lds;
    asm volatile("global_load_async_to_lds_b128 %0, %1, off"
                 :: "v"(l), "v"(src) : "memory");
}
__device__ __forceinline__ void wait_async0() {
    asm volatile("s_wait_asynccnt 0x0" ::: "memory");
}

// ---------------------------------------------------------------------------
// Pass 1: per-row branch multipliers (outs[:, -1] dot + h[:, -1] signs).
// ---------------------------------------------------------------------------
__global__ __launch_bounds__(256) void mask_kernel(
    const float* __restrict__ x, const float* __restrict__ h,
    const float* __restrict__ Wih,
    float* __restrict__ mf, float* __restrict__ mi, float* __restrict__ mp) {
    const int lane = threadIdx.x & 31;
    const int wv = threadIdx.x >> 5;
    const int b = blockIdx.x * 8 + wv;

    const float* __restrict__ wrow = Wih + (size_t)(NDIM - 1) * KDIM;
    const float* __restrict__ xrow = x + (size_t)b * KDIM;

    float s = 0.0f;
    for (int k = lane; k < KDIM; k += 32)
        s = fmaf(xrow[k], wrow[k], s);
    #pragma unroll
    for (int off = 16; off > 0; off >>= 1)
        s += __shfl_xor(s, off, 32);

    if (lane == 0) {
        const bool nt = (s >= -1.0f);
        const float hl = h[(size_t)b * NDIM + (NDIM - 1)];
        mf[b] = (nt && hl > 0.0f) ? 1.0f : 0.0f;
        mi[b] = (nt && hl < 0.0f) ? 1.0f : 0.0f;
        mp[b] = nt ? 0.0f : 1.0f;
    }
}

// ---------------------------------------------------------------------------
// Pass 2a: elementwise hi/lo split for K-contiguous matrices (x, h, Wih).
// ---------------------------------------------------------------------------
__global__ __launch_bounds__(256) void conv_direct_kernel(
    const float* __restrict__ src,
    unsigned short* __restrict__ dhi, unsigned short* __restrict__ dlo) {
    const size_t c = (size_t)blockIdx.x * 256 + threadIdx.x;  // float4 index
    const float4 v = *(const float4*)(src + c * 4);
    unsigned short h0, h1, h2, h3, l0, l1, l2, l3;
    split_bf16(v.x, h0, l0); split_bf16(v.y, h1, l1);
    split_bf16(v.z, h2, l2); split_bf16(v.w, h3, l3);
    uint2 ph, pl;
    ph.x = (unsigned)h0 | ((unsigned)h1 << 16);
    ph.y = (unsigned)h2 | ((unsigned)h3 << 16);
    pl.x = (unsigned)l0 | ((unsigned)l1 << 16);
    pl.y = (unsigned)l2 | ((unsigned)l3 << 16);
    *(uint2*)(dhi + c * 4) = ph;
    *(uint2*)(dlo + c * 4) = pl;
}

// ---------------------------------------------------------------------------
// Pass 2b: Whh / Winv [K][N] -> transpose-convert to [N][K] bf16 hi/lo.
// ---------------------------------------------------------------------------
__global__ __launch_bounds__(256) void transpose_conv_kernel(
    const float* __restrict__ s0, const float* __restrict__ s1,
    unsigned short* __restrict__ d0hi, unsigned short* __restrict__ d0lo,
    unsigned short* __restrict__ d1hi, unsigned short* __restrict__ d1lo) {
    __shared__ unsigned short thi[64 * 80];   // [n][k], 160B rows
    __shared__ unsigned short tlo[64 * 80];

    const float* __restrict__ src = blockIdx.z ? s1 : s0;
    unsigned short* __restrict__ dhi = blockIdx.z ? d1hi : d0hi;
    unsigned short* __restrict__ dlo = blockIdx.z ? d1lo : d0lo;

    const int k0 = blockIdx.x * 64;
    const int n0 = blockIdx.y * 64;
    const int tid = threadIdx.x;

    #pragma unroll
    for (int i = 0; i < 4; ++i) {
        const int c = tid + i * 256;
        const int kr = c >> 4;
        const int nq = c & 15;
        const float4 v = *(const float4*)(src + (size_t)(k0 + kr) * NDIM + n0 + nq * 4);
        unsigned short hh, ll;
        split_bf16(v.x, hh, ll); thi[(nq * 4 + 0) * 80 + kr] = hh; tlo[(nq * 4 + 0) * 80 + kr] = ll;
        split_bf16(v.y, hh, ll); thi[(nq * 4 + 1) * 80 + kr] = hh; tlo[(nq * 4 + 1) * 80 + kr] = ll;
        split_bf16(v.z, hh, ll); thi[(nq * 4 + 2) * 80 + kr] = hh; tlo[(nq * 4 + 2) * 80 + kr] = ll;
        split_bf16(v.w, hh, ll); thi[(nq * 4 + 3) * 80 + kr] = hh; tlo[(nq * 4 + 3) * 80 + kr] = ll;
    }
    __syncthreads();
    #pragma unroll
    for (int i = 0; i < 2; ++i) {
        const int c = tid + i * 256;
        const int n = c >> 3;
        const int kq = c & 7;
        const uint4 vh = *(const uint4*)(thi + n * 80 + kq * 8);
        const uint4 vl = *(const uint4*)(tlo + n * 80 + kq * 8);
        *(uint4*)(dhi + (size_t)(n0 + n) * KDIM + k0 + kq * 8) = vh;
        *(uint4*)(dlo + (size_t)(n0 + n) * KDIM + k0 + kq * 8) = vl;
    }
}

// ---------------------------------------------------------------------------
// Pass 3: fused triple GEMM (bf16x3 split, f32 accumulate) + blend epilogue.
// All operands pre-split to bf16 planes -> staging is pure async 16B copies.
// Double-buffered LDS, one barrier per K-block, WMMA overlapped with copies.
// ---------------------------------------------------------------------------
struct GemmArgs {
    const unsigned short *xhi, *xlo, *hhi, *hlo;
    const unsigned short *w1h, *w1l, *w2h, *w2l, *w3h, *w3l;
};

__device__ __forceinline__ void stage_async(
    unsigned short* sb, const GemmArgs& g, int m0, int n0, int kb, int tid) {
    #pragma unroll
    for (int i = 0; i < 2; ++i) {
        const int c = tid + i * 256;            // 512 chunks per A plane
        const int row = c >> 2;                 // [0,128)
        const int kc = (c & 3) * 8;             // k offset
        const size_t go = (size_t)(m0 + row) * KDIM + kb + kc;
        const int l = row * SAB + kc;
        cp16(sb + OXH + l, g.xhi + go);
        cp16(sb + OXL + l, g.xlo + go);
        cp16(sb + OHH + l, g.hhi + go);
        cp16(sb + OHL + l, g.hlo + go);
    }
    {
        const int n = tid >> 2;                 // [0,64)
        const int kc = (tid & 3) * 8;
        const size_t go = (size_t)(n0 + n) * KDIM + kb + kc;
        const int l = n * SBB + kc;
        cp16(sb + OB1H + l, g.w1h + go);
        cp16(sb + OB1L + l, g.w1l + go);
        cp16(sb + OB2H + l, g.w2h + go);
        cp16(sb + OB2L + l, g.w2l + go);
        cp16(sb + OB3H + l, g.w3h + go);
        cp16(sb + OB3L + l, g.w3l + go);
    }
}

__device__ __forceinline__ void compute_block(
    const unsigned short* sb, v8f (&acc)[3][2][2],
    int wm, int wn, int lm, int half) {
    // A(16x32) lanes: half 0 -> K 0..7 & 16..23 ; half 1 -> K 8..15 & 24..31
    v16bf axh[2], axl[2], ahh[2], ahl[2];
    #pragma unroll
    for (int i = 0; i < 2; ++i) {
        const int r = (wm + i * 16 + lm) * SAB;
        const int c0 = r + 8 * half;
        const int c1 = r + 16 + 8 * half;
        axh[i] = combine16(*(const v8bf*)(sb + OXH + c0), *(const v8bf*)(sb + OXH + c1));
        axl[i] = combine16(*(const v8bf*)(sb + OXL + c0), *(const v8bf*)(sb + OXL + c1));
        ahh[i] = combine16(*(const v8bf*)(sb + OHH + c0), *(const v8bf*)(sb + OHH + c1));
        ahl[i] = combine16(*(const v8bf*)(sb + OHL + c0), *(const v8bf*)(sb + OHL + c1));
    }
    // B(32x16) lanes: half 0 -> K 0..7 & 16..23 ; half 1 -> K 8..15 & 24..31
    #pragma unroll
    for (int j = 0; j < 2; ++j) {
        const int r = (wn + j * 16 + lm) * SBB + 8 * half;
        const v16bf f1h = combine16(*(const v8bf*)(sb + OB1H + r), *(const v8bf*)(sb + OB1H + r + 16));
        const v16bf f1l = combine16(*(const v8bf*)(sb + OB1L + r), *(const v8bf*)(sb + OB1L + r + 16));
        const v16bf f2h = combine16(*(const v8bf*)(sb + OB2H + r), *(const v8bf*)(sb + OB2H + r + 16));
        const v16bf f2l = combine16(*(const v8bf*)(sb + OB2L + r), *(const v8bf*)(sb + OB2L + r + 16));
        const v16bf f3h = combine16(*(const v8bf*)(sb + OB3H + r), *(const v8bf*)(sb + OB3H + r + 16));
        const v16bf f3l = combine16(*(const v8bf*)(sb + OB3L + r), *(const v8bf*)(sb + OB3L + r + 16));
        #pragma unroll
        for (int i = 0; i < 2; ++i) {
            acc[0][i][j] = wmma_bf16(axh[i], f1h, acc[0][i][j]);
            acc[0][i][j] = wmma_bf16(axh[i], f1l, acc[0][i][j]);
            acc[0][i][j] = wmma_bf16(axl[i], f1h, acc[0][i][j]);
            acc[1][i][j] = wmma_bf16(ahh[i], f2h, acc[1][i][j]);
            acc[1][i][j] = wmma_bf16(ahh[i], f2l, acc[1][i][j]);
            acc[1][i][j] = wmma_bf16(ahl[i], f2h, acc[1][i][j]);
            acc[2][i][j] = wmma_bf16(ahh[i], f3h, acc[2][i][j]);
            acc[2][i][j] = wmma_bf16(ahh[i], f3l, acc[2][i][j]);
            acc[2][i][j] = wmma_bf16(ahl[i], f3h, acc[2][i][j]);
        }
    }
}

__global__ __launch_bounds__(256) void fused_gemm_kernel(
    GemmArgs g, const float* __restrict__ h,
    const float* __restrict__ mf, const float* __restrict__ mi,
    const float* __restrict__ mp, float* __restrict__ out) {
    __shared__ unsigned short smem[2 * BUFE];   // 2 x 71680 B

    const int tid = threadIdx.x;
    const int lane = tid & 31;
    const int wv = tid >> 5;
    const int wm = (wv & 3) * 32;
    const int wn = (wv >> 2) * 32;
    const int m0 = blockIdx.y * BM;
    const int n0 = blockIdx.x * BN;
    const int lm = lane & 15;
    const int half = lane >> 4;

    unsigned short* s0 = smem;
    unsigned short* s1 = smem + BUFE;

    v8f acc[3][2][2] = {};

    stage_async(s0, g, m0, n0, 0, tid);
    wait_async0();
    __syncthreads();

    for (int kb = 0; kb < KDIM; kb += 2 * BK) {
        stage_async(s1, g, m0, n0, kb + BK, tid);      // always valid (kb+BK <= K-BK)
        compute_block(s0, acc, wm, wn, lm, half);
        wait_async0();
        __syncthreads();

        if (kb + 2 * BK < KDIM)
            stage_async(s0, g, m0, n0, kb + 2 * BK, tid);
        compute_block(s1, acc, wm, wn, lm, half);
        wait_async0();
        __syncthreads();
    }

    // ---- epilogue ----
    #pragma unroll
    for (int i = 0; i < 2; ++i) {
        #pragma unroll
        for (int v = 0; v < 8; ++v) {
            const int m = m0 + wm + i * 16 + (half << 3) + v;
            const float fm = mf[m];
            const float fi = mi[m];
            const float fp = mp[m];
            const float* __restrict__ hrow = h + (size_t)m * NDIM;
            float* __restrict__ orow = out + (size_t)m * NDIM;
            #pragma unroll
            for (int j = 0; j < 2; ++j) {
                const int n = n0 + wn + j * 16 + lm;
                orow[n] = acc[0][i][j][v] + fm * acc[1][i][j][v] +
                          fi * acc[2][i][j][v] + fp * hrow[n];
            }
        }
    }
}

extern "C" void kernel_launch(void* const* d_in, const int* in_sizes, int n_in,
                              void* d_out, int out_size, void* d_ws, size_t ws_size,
                              hipStream_t stream) {
    const float* x    = (const float*)d_in[0];
    const float* h    = (const float*)d_in[1];
    const float* Wih  = (const float*)d_in[2];
    const float* Whh  = (const float*)d_in[3];
    const float* Winv = (const float*)d_in[4];
    float* out = (float*)d_out;

    const size_t MK = (size_t)MDIM * KDIM;          // 33.5M elems
    const size_t NK = (size_t)NDIM * KDIM;          // 4.2M elems
    unsigned short* p = (unsigned short*)d_ws;
    GemmArgs g;
    unsigned short* xhi = p;             unsigned short* xlo = p + MK;
    unsigned short* hhi = p + 2 * MK;    unsigned short* hlo = p + 3 * MK;
    unsigned short* wp = p + 4 * MK;
    g.xhi = xhi; g.xlo = xlo; g.hhi = hhi; g.hlo = hlo;
    g.w1h = wp;          g.w1l = wp + NK;
    g.w2h = wp + 2 * NK; g.w2l = wp + 3 * NK;
    g.w3h = wp + 4 * NK; g.w3l = wp + 5 * NK;
    float* mf = (float*)(wp + 6 * NK);
    float* mi = mf + MDIM;
    float* mp = mi + MDIM;

    mask_kernel<<<MDIM / 8, 256, 0, stream>>>(x, h, Wih, mf, mi, mp);

    conv_direct_kernel<<<(unsigned)(MK / 1024), 256, 0, stream>>>(x, xhi, xlo);
    conv_direct_kernel<<<(unsigned)(MK / 1024), 256, 0, stream>>>(h, hhi, hlo);
    conv_direct_kernel<<<(unsigned)(NK / 1024), 256, 0, stream>>>(Wih, (unsigned short*)g.w1h, (unsigned short*)g.w1l);

    dim3 tgrid(KDIM / 64, NDIM / 64, 2);
    transpose_conv_kernel<<<tgrid, 256, 0, stream>>>(
        Whh, Winv,
        (unsigned short*)g.w2h, (unsigned short*)g.w2l,
        (unsigned short*)g.w3h, (unsigned short*)g.w3l);

    dim3 grid(NDIM / BN, MDIM / BM);                // 32 x 128 workgroups
    fused_gemm_kernel<<<grid, 256, 0, stream>>>(g, h, mf, mi, mp, out);
}
